// ExampleModel_40939628265765
// MI455X (gfx1250) — matrix operand order repaired and checked by reading, not compiled, not run on hardware
//
#include <hip/hip_runtime.h>
#include <hip/hip_bf16.h>
#include <math.h>

// ---------------------------------------------------------------------------
// Problem constants (match reference)
// ---------------------------------------------------------------------------
static constexpr int Bn = 4, Tn = 2048, Dn = 2048, Hn = 2048, En = 8, Kn = 2;
static constexpr int Sn = Bn * Tn;          // 8192 tokens
static constexpr int Cap = 2048;            // per-expert capacity

// ---------------------------------------------------------------------------
// CDNA5 async global->LDS path (guarded; sync fallback keeps compile green)
// Builtin signature (from hipcc diagnostic): (int4 AS1*, int4 AS3*, imm, imm)
// ---------------------------------------------------------------------------
#if defined(__has_builtin)
#if __has_builtin(__builtin_amdgcn_global_load_async_to_lds_b128)
#define MOE_ASYNC_LDS 1
#endif
#endif
#ifndef MOE_ASYNC_LDS
#define MOE_ASYNC_LDS 0
#endif

#if MOE_ASYNC_LDS
typedef int moe_v4i __attribute__((vector_size(4 * sizeof(int))));
typedef __attribute__((address_space(1))) moe_v4i* moe_gp1;
typedef __attribute__((address_space(3))) moe_v4i* moe_lp3;
#define MOE_ASYNC_B128(gsrc, ldst)                                             \
  __builtin_amdgcn_global_load_async_to_lds_b128((moe_gp1)(void*)(gsrc),       \
                                                 (moe_lp3)(void*)(ldst), 0, 0)
#endif

__device__ inline void moe_wait_async_copies() {
#if MOE_ASYNC_LDS
#if __has_builtin(__builtin_amdgcn_s_wait_asynccnt)
  __builtin_amdgcn_s_wait_asynccnt(0);
#else
  asm volatile("s_wait_asynccnt 0x0" ::: "memory");
#endif
#endif
}

// ---------------------------------------------------------------------------
// WMMA vector types (gfx1250, wave32)
// ---------------------------------------------------------------------------
typedef __attribute__((ext_vector_type(16))) __bf16 v16bf;
typedef __attribute__((ext_vector_type(8)))  __bf16 bf16x8;
typedef __attribute__((ext_vector_type(4)))  __bf16 bf16x4;
typedef __attribute__((ext_vector_type(8)))  float  v8f;

union FragBF16 { v16bf v; bf16x8 h[2]; };

// ---------------------------------------------------------------------------
// 1) f32 -> bf16 convert + transpose of last two dims: in [E,R,Cc] -> out [E,Cc,R]
// ---------------------------------------------------------------------------
__global__ void __launch_bounds__(256)
moe_cvt_transpose(const float* __restrict__ in, __bf16* __restrict__ out,
                  int R, int Cc) {
  __shared__ float tile[32][33];
  const int e = blockIdx.z;
  const float* pin = in + (size_t)e * R * Cc;
  __bf16* pout = out + (size_t)e * R * Cc;
  const int c0 = blockIdx.x * 32, r0 = blockIdx.y * 32;
  const int tx = threadIdx.x, ty = threadIdx.y;   // block (32, 8)
#pragma unroll
  for (int i = 0; i < 32; i += 8)
    tile[ty + i][tx] = pin[(size_t)(r0 + ty + i) * Cc + c0 + tx];
  __syncthreads();
#pragma unroll
  for (int i = 0; i < 32; i += 8)
    pout[(size_t)(c0 + ty + i) * R + r0 + tx] = (__bf16)tile[tx][ty + i];
}

// ---------------------------------------------------------------------------
// 2) Gating: logits = x[s] @ wg, top-2 (ties -> lower index, like lax.top_k),
//    normalized pair weights w_i = p_i / (p0 + p1). One wave per token.
// ---------------------------------------------------------------------------
__global__ void __launch_bounds__(256)
moe_gate_topk(const float* __restrict__ x, const float* __restrict__ wg,
              int* __restrict__ eidx, float* __restrict__ wnorm) {
  const int wave = threadIdx.x >> 5, lane = threadIdx.x & 31;
  const int s = blockIdx.x * 8 + wave;
  const float* xs = x + (size_t)s * Dn;
  float acc[En];
#pragma unroll
  for (int e = 0; e < En; ++e) acc[e] = 0.f;
  for (int d = lane; d < Dn; d += 32) {
    const float xv = xs[d];
    const float* wr = wg + (size_t)d * En;
#pragma unroll
    for (int e = 0; e < En; ++e) acc[e] += xv * wr[e];
  }
#pragma unroll
  for (int e = 0; e < En; ++e)
#pragma unroll
    for (int off = 16; off > 0; off >>= 1)
      acc[e] += __shfl_xor(acc[e], off, 32);
  if (lane == 0) {
    int i0 = 0;
#pragma unroll
    for (int e = 1; e < En; ++e) if (acc[e] > acc[i0]) i0 = e;
    int i1 = -1;
#pragma unroll
    for (int e = 0; e < En; ++e)
      if (e != i0 && (i1 < 0 || acc[e] > acc[i1])) i1 = e;
    const float p1 = __expf(acc[i1] - acc[i0]);   // p0 = 1
    const float inv = 1.f / (1.f + p1);
    eidx[s * 2 + 0] = i0;  wnorm[s * 2 + 0] = inv;
    eidx[s * 2 + 1] = i1;  wnorm[s * 2 + 1] = p1 * inv;
  }
}

// ---------------------------------------------------------------------------
// 3) slot2token fill
// ---------------------------------------------------------------------------
__global__ void __launch_bounds__(256)
moe_fill_neg1(int* __restrict__ p, int n) {
  const int i = blockIdx.x * 256 + threadIdx.x;
  if (i < n) p[i] = -1;
}

// ---------------------------------------------------------------------------
// 4) Capacity-aware slot assignment: loc[k][s] = offs[k][e] + rank_within(k,e,s)
// ---------------------------------------------------------------------------
__global__ void __launch_bounds__(256)
moe_scan_assign(const int* __restrict__ eidx, int* __restrict__ locarr,
                int* __restrict__ slot2tok) {
  __shared__ int cnt[256][16];
  __shared__ int totals[16];
  __shared__ int offsArr[16];
  const int tid = threadIdx.x;
  const int chunk = Sn / 256;          // 32 tokens/thread
  const int s0 = tid * chunk;
  int c[16];
#pragma unroll
  for (int j = 0; j < 16; ++j) c[j] = 0;
  for (int i = 0; i < chunk; ++i) {
    const int s = s0 + i;
    c[0 + eidx[s * 2 + 0]]++;
    c[8 + eidx[s * 2 + 1]]++;
  }
#pragma unroll
  for (int j = 0; j < 16; ++j) cnt[tid][j] = c[j];
  __syncthreads();
  if (tid < 16) {                      // exclusive scan across thread chunks
    int run = 0;
    for (int t = 0; t < 256; ++t) { const int v = cnt[t][tid]; cnt[t][tid] = run; run += v; }
    totals[tid] = run;
  }
  __syncthreads();
  if (tid < 16) offsArr[tid] = (tid < 8) ? 0 : totals[tid - 8];  // offs[1][e]=counts[0][e]
  __syncthreads();
#pragma unroll
  for (int j = 0; j < 16; ++j) c[j] = cnt[tid][j];
  for (int i = 0; i < chunk; ++i) {
    const int s = s0 + i;
#pragma unroll
    for (int k = 0; k < Kn; ++k) {
      const int e = eidx[s * 2 + k];
      const int ke = k * 8 + e;
      const int loc = offsArr[ke] + c[ke];
      c[ke]++;
      locarr[s * 2 + k] = loc;
      if (loc < Cap) slot2tok[e * Cap + loc] = s;
    }
  }
}

// ---------------------------------------------------------------------------
// 5) Dispatch: gather token rows into bf16 expert buffers disp[E,C,D]
// ---------------------------------------------------------------------------
__global__ void __launch_bounds__(256)
moe_dispatch(const float* __restrict__ x, const int* __restrict__ slot2tok,
             __bf16* __restrict__ disp) {
  const int slot = blockIdx.x;               // e*Cap + c
  const int s = slot2tok[slot];
  __bf16* drow = disp + (size_t)slot * Dn;
  const int d = threadIdx.x * 4;
  if (s >= 0) {
    const float* xr = x + (size_t)s * Dn;
    for (int base = 0; base < Dn; base += 1024) {
      const float4 v = *(const float4*)(xr + base + d);
      bf16x4 o;
      o[0] = (__bf16)v.x; o[1] = (__bf16)v.y; o[2] = (__bf16)v.z; o[3] = (__bf16)v.w;
      *(bf16x4*)(drow + base + d) = o;
    }
  } else {
    const bf16x4 z = {};
    for (int base = 0; base < Dn; base += 1024)
      *(bf16x4*)(drow + base + d) = z;
  }
}

// ---------------------------------------------------------------------------
// 6/7) Grouped expert GEMM, LDS double-buffered, async global->LDS staging.
//      C_out = act(A[E,M,K] @ Bt[E,N,K]^T + bias[E,N])
//      Block = 256 thr (8 waves), tile 128x128, wave tile 64x32 (4x2 WMMA frags)
//      MODE 1: ReLU + bf16 out ; MODE 0: f32 out
// ---------------------------------------------------------------------------
static constexpr int TPAD = 40;                  // padded LDS row stride (elems)
static constexpr int TILE_ELEMS = 128 * TPAD;    // one 128x32 tile (padded)

template <int MODE>
__global__ void __launch_bounds__(256)
moe_gemm_bf16(const __bf16* __restrict__ A, const __bf16* __restrict__ Bt,
              const float* __restrict__ bias, void* __restrict__ Out,
              int M, int N, int Kd) {
  __shared__ __bf16 lds[2 * 2 * TILE_ELEMS];     // [buf][A|B][128][TPAD]

  const int e = blockIdx.z;
  const __bf16* Ag = A  + (size_t)e * M * Kd + (size_t)blockIdx.y * 128 * Kd;
  const __bf16* Bg = Bt + (size_t)e * N * Kd + (size_t)blockIdx.x * 128 * Kd;

  const int tid  = threadIdx.x;
  const int wave = tid >> 5;
  const int lane = tid & 31;
  const int half = lane >> 4;
  const int l16  = lane & 15;
  const int wm   = (wave & 1) * 64;   // wave M offset within tile
  const int wn   = (wave >> 1) * 32;  // wave N offset within tile

  // staging map: 512 x 16B chunks per tile; thread moves rows r0c and r0c+64
  const int r0c = tid >> 2;
  const int c0c = (tid & 3) * 8;      // element offset of 16B chunk in row

  auto stage = [&](int k0, int buf) {
    __bf16* As = lds + buf * (2 * TILE_ELEMS);
    __bf16* Bs = As + TILE_ELEMS;
#if MOE_ASYNC_LDS
#pragma unroll
    for (int j = 0; j < 2; ++j) {
      const int r = r0c + j * 64;
      MOE_ASYNC_B128(Ag + (size_t)r * Kd + k0 + c0c, As + r * TPAD + c0c);
      MOE_ASYNC_B128(Bg + (size_t)r * Kd + k0 + c0c, Bs + r * TPAD + c0c);
    }
#else
#pragma unroll
    for (int j = 0; j < 2; ++j) {
      const int r = r0c + j * 64;
      *(bf16x8*)(As + r * TPAD + c0c) = *(const bf16x8*)(Ag + (size_t)r * Kd + k0 + c0c);
      *(bf16x8*)(Bs + r * TPAD + c0c) = *(const bf16x8*)(Bg + (size_t)r * Kd + k0 + c0c);
    }
#endif
  };

  v8f acc[4][2] = {};

  stage(0, 0);
  moe_wait_async_copies();
  __syncthreads();

  for (int k0 = 0; k0 < Kd; k0 += 32) {
    const int p = (k0 >> 5) & 1;
    if (k0 + 32 < Kd) stage(k0 + 32, p ^ 1);

    const __bf16* As = lds + p * (2 * TILE_ELEMS);
    const __bf16* Bs = As + TILE_ELEMS;

    v16bf a[4], b[2];
#pragma unroll
    for (int i = 0; i < 4; ++i) {       // A 16x32 frag: 2x16B, rows conflict-free
      const int row = wm + i * 16 + l16;
      FragBF16 f;
      f.h[0] = *(const bf16x8*)(As + row * TPAD + half * 8);
      f.h[1] = *(const bf16x8*)(As + row * TPAD + 16 + half * 8);
      a[i] = f.v;
    }
#pragma unroll
    for (int j = 0; j < 2; ++j) {       // B 32x16 frag: 2x16B per column, same k order as A
      const int col = wn + j * 16 + l16;
      FragBF16 f;
      f.h[0] = *(const bf16x8*)(Bs + col * TPAD + half * 8);
      f.h[1] = *(const bf16x8*)(Bs + col * TPAD + 16 + half * 8);
      b[j] = f.v;
    }
#pragma unroll
    for (int i = 0; i < 4; ++i)
#pragma unroll
      for (int j = 0; j < 2; ++j)
        acc[i][j] = __builtin_amdgcn_wmma_f32_16x16x32_bf16(
            false, a[i], false, b[j], (short)0, acc[i][j], false, false);

    moe_wait_async_copies();   // next buffer landed
    __syncthreads();           // all waves done reading buffer p
  }

  // C/D layout: VGPR r -> row = (half?8:0)+r ; col = lane&15
#pragma unroll
  for (int i = 0; i < 4; ++i) {
#pragma unroll
    for (int j = 0; j < 2; ++j) {
      const int col = blockIdx.x * 128 + wn + j * 16 + l16;
      const float bv = bias[(size_t)e * N + col];
#pragma unroll
      for (int r = 0; r < 8; ++r) {
        const int row = blockIdx.y * 128 + wm + i * 16 + half * 8 + r;
        const float v = acc[i][j][r] + bv;
        if (MODE == 1) {
          ((__bf16*)Out)[((size_t)e * M + row) * N + col] = (__bf16)fmaxf(v, 0.f);
        } else {
          ((float*)Out)[((size_t)e * M + row) * N + col] = v;
        }
      }
    }
  }
}

// ---------------------------------------------------------------------------
// 8) Combine + row-sum over D: z[s] = sum_d( sum_k w_k * ye[e_k, loc_k, d] )
// ---------------------------------------------------------------------------
__global__ void __launch_bounds__(256)
moe_combine(const float* __restrict__ ye, const int* __restrict__ eidx,
            const int* __restrict__ locarr, const float* __restrict__ wnorm,
            float* __restrict__ z) {
  const int s = blockIdx.x;
  const int e0 = eidx[s * 2 + 0], e1 = eidx[s * 2 + 1];
  int l0 = locarr[s * 2 + 0], l1 = locarr[s * 2 + 1];
  const float w0 = (l0 < Cap) ? wnorm[s * 2 + 0] : 0.f;
  const float w1 = (l1 < Cap) ? wnorm[s * 2 + 1] : 0.f;
  l0 = l0 < Cap ? l0 : Cap - 1;
  l1 = l1 < Cap ? l1 : Cap - 1;
  const float* r0 = ye + ((size_t)e0 * Cap + l0) * Dn;
  const float* r1 = ye + ((size_t)e1 * Cap + l1) * Dn;
  float p = 0.f;
  for (int d = threadIdx.x; d < Dn; d += 256) p += w0 * r0[d] + w1 * r1[d];
  __shared__ float red[256];
  red[threadIdx.x] = p;
  __syncthreads();
#pragma unroll
  for (int off = 128; off > 0; off >>= 1) {
    if (threadIdx.x < off) red[threadIdx.x] += red[threadIdx.x + off];
    __syncthreads();
  }
  if (threadIdx.x == 0) z[s] = red[0];
}

// ---------------------------------------------------------------------------
// 9) log_softmax over T per batch row
// ---------------------------------------------------------------------------
__global__ void __launch_bounds__(256)
moe_logsoftmax(const float* __restrict__ z, float* __restrict__ out) {
  const int b = blockIdx.x;
  const float* zr = z + (size_t)b * Tn;
  __shared__ float red[256];
  float m = -INFINITY;
  for (int t = threadIdx.x; t < Tn; t += 256) m = fmaxf(m, zr[t]);
  red[threadIdx.x] = m;
  __syncthreads();
#pragma unroll
  for (int off = 128; off > 0; off >>= 1) {
    if (threadIdx.x < off)
      red[threadIdx.x] = fmaxf(red[threadIdx.x], red[threadIdx.x + off]);
    __syncthreads();
  }
  m = red[0];
  __syncthreads();
  float ssum = 0.f;
  for (int t = threadIdx.x; t < Tn; t += 256) ssum += __expf(zr[t] - m);
  red[threadIdx.x] = ssum;
  __syncthreads();
#pragma unroll
  for (int off = 128; off > 0; off >>= 1) {
    if (threadIdx.x < off) red[threadIdx.x] += red[threadIdx.x + off];
    __syncthreads();
  }
  const float lse = m + logf(red[0]);
  for (int t = threadIdx.x; t < Tn; t += 256)
    out[(size_t)b * Tn + t] = zr[t] - lse;
}

// ---------------------------------------------------------------------------
// Host launch
// ---------------------------------------------------------------------------
extern "C" void kernel_launch(void* const* d_in, const int* in_sizes, int n_in,
                              void* d_out, int out_size, void* d_ws, size_t ws_size,
                              hipStream_t stream) {
  (void)in_sizes; (void)n_in; (void)out_size; (void)ws_size;

  const float* x  = (const float*)d_in[0];   // [B,T,D]
  const float* wg = (const float*)d_in[1];   // [D,E]
  const float* w1 = (const float*)d_in[2];   // [E,D,H]
  const float* b1 = (const float*)d_in[3];   // [E,H]
  const float* w2 = (const float*)d_in[4];   // [E,H,D]
  const float* b2 = (const float*)d_in[5];   // [E,D]
  float* out = (float*)d_out;                // [B,T]

  // workspace carve-up (256B aligned)
  char* ws = (char*)d_ws;
  size_t off = 0;
  auto carve = [&](size_t bytes) -> void* {
    void* p = ws + off;
    off = (off + bytes + 255) & ~(size_t)255;
    return p;
  };
  const size_t edh = (size_t)En * Dn * Hn;         // 33.5M elems
  __bf16* w1t   = (__bf16*)carve(edh * 2);         // [E,H,D] bf16
  __bf16* w2t   = (__bf16*)carve(edh * 2);         // [E,D,H] bf16
  __bf16* disp  = (__bf16*)carve((size_t)En * Cap * Dn * 2);   // [E,C,D] bf16
  __bf16* hbuf  = (__bf16*)carve((size_t)En * Cap * Hn * 2);   // [E,C,H] bf16
  float*  ye    = (float*)carve((size_t)En * Cap * Dn * 4);    // [E,C,D] f32
  int*    eidx  = (int*)carve((size_t)Sn * Kn * 4);
  float*  wn    = (float*)carve((size_t)Sn * Kn * 4);
  int*    loca  = (int*)carve((size_t)Sn * Kn * 4);
  int*    s2t   = (int*)carve((size_t)En * Cap * 4);
  float*  zbuf  = (float*)carve((size_t)Sn * 4);

  // 1) weight convert + transpose (bf16 weights: 128MB, L2-resident on 192MB L2)
  {
    dim3 blk(32, 8);
    dim3 g1(Hn / 32, Dn / 32, En);
    moe_cvt_transpose<<<g1, blk, 0, stream>>>(w1, w1t, Dn, Hn);
    dim3 g2(Dn / 32, Hn / 32, En);
    moe_cvt_transpose<<<g2, blk, 0, stream>>>(w2, w2t, Hn, Dn);
  }
  // 2) gating
  moe_gate_topk<<<dim3(Sn / 8), dim3(256), 0, stream>>>(x, wg, eidx, wn);
  // 3) slot2token init
  moe_fill_neg1<<<dim3((En * Cap + 255) / 256), dim3(256), 0, stream>>>(s2t, En * Cap);
  // 4) capacity scan / slot assignment
  moe_scan_assign<<<dim3(1), dim3(256), 0, stream>>>(eidx, loca, s2t);
  // 5) dispatch
  moe_dispatch<<<dim3(En * Cap), dim3(256), 0, stream>>>(x, s2t, disp);
  // 6) h = relu(disp @ w1 + b1)  (bf16 out)
  moe_gemm_bf16<1><<<dim3(Hn / 128, Cap / 128, En), dim3(256), 0, stream>>>(
      disp, w1t, b1, (void*)hbuf, Cap, Hn, Dn);
  // 7) ye = h @ w2 + b2          (f32 out)
  moe_gemm_bf16<0><<<dim3(Dn / 128, Cap / 128, En), dim3(256), 0, stream>>>(
      hbuf, w2t, b2, (void*)ye, Cap, Dn, Hn);
  // 8) combine + sum over D
  moe_combine<<<dim3(Sn), dim3(256), 0, stream>>>(ye, eidx, loca, wn, zbuf);
  // 9) log_softmax over T
  moe_logsoftmax<<<dim3(Bn), dim3(256), 0, stream>>>(zbuf, out);
}
